// GAT_degree_87479893885519
// MI455X (gfx1250) — compile-verified, matrix operand order checked by reading, not run on hardware
//
#include <hip/hip_runtime.h>

typedef __attribute__((ext_vector_type(2)))  float    v2f;
typedef __attribute__((ext_vector_type(8)))  float    v8f;
typedef __attribute__((ext_vector_type(8)))  _Float16 v8h;
typedef __attribute__((ext_vector_type(16))) _Float16 v16h;

#define NN 4096   // nodes
#define NEGBIG (-1.0e30f)

// ============================================================================
// Kernel 1: Wh = X @ W (one 16x64 tile per wave, per head) using FP32 WMMA
//   16x16x4.  Also emits transposed f16 copy WhT[h][o][n] (m contiguous) so
//   the flash-attn B operand maps to contiguous 16B loads.
// ============================================================================
__global__ void __launch_bounds__(32)
gat_gemm_wh(const float* __restrict__ X,   // (NN, F)
            const float* __restrict__ W,   // (H, F, 64)
            float* __restrict__ Wh,        // (H, NN, 64)
            _Float16* __restrict__ WhT,    // (H, 64, NN)
            int F)
{
  const int nt   = blockIdx.x;
  const int h    = blockIdx.y;
  const int lane = threadIdx.x;
  const int M    = lane & 15;
  const int hi   = lane >> 4;
  const int row  = nt * 16 + M;

  const float* xrow = X + (size_t)row * F;
  const float* Wb   = W + (size_t)h * F * 64;

  v8f acc0 = {}, acc1 = {}, acc2 = {}, acc3 = {};

  for (int k = 0; k < F; k += 4) {
    const int k0 = k + 2 * hi;
    v2f a;
    a.x = xrow[k0];
    a.y = xrow[k0 + 1];
    const float* w0 = Wb + (size_t)k0 * 64;
    const float* w1 = w0 + 64;
    v2f b0; b0.x = w0[ 0 + M]; b0.y = w1[ 0 + M];
    v2f b1; b1.x = w0[16 + M]; b1.y = w1[16 + M];
    v2f b2; b2.x = w0[32 + M]; b2.y = w1[32 + M];
    v2f b3; b3.x = w0[48 + M]; b3.y = w1[48 + M];
    acc0 = __builtin_amdgcn_wmma_f32_16x16x4_f32(false, a, false, b0, (short)0, acc0, false, false);
    acc1 = __builtin_amdgcn_wmma_f32_16x16x4_f32(false, a, false, b1, (short)0, acc1, false, false);
    acc2 = __builtin_amdgcn_wmma_f32_16x16x4_f32(false, a, false, b2, (short)0, acc2, false, false);
    acc3 = __builtin_amdgcn_wmma_f32_16x16x4_f32(false, a, false, b3, (short)0, acc3, false, false);
  }

  float*    WhH = Wh  + (size_t)h * NN * 64;
  _Float16* WtH = WhT + (size_t)h * 64 * NN;
  #pragma unroll
  for (int c = 0; c < 8; ++c) {
    const int orow = nt * 16 + c + 8 * hi;
    WhH[(size_t)orow * 64 +  0 + M] = acc0[c];
    WhH[(size_t)orow * 64 + 16 + M] = acc1[c];
    WhH[(size_t)orow * 64 + 32 + M] = acc2[c];
    WhH[(size_t)orow * 64 + 48 + M] = acc3[c];
    WtH[(size_t)( 0 + M) * NN + orow] = (_Float16)acc0[c];
    WtH[(size_t)(16 + M) * NN + orow] = (_Float16)acc1[c];
    WtH[(size_t)(32 + M) * NN + orow] = (_Float16)acc2[c];
    WtH[(size_t)(48 + M) * NN + orow] = (_Float16)acc3[c];
  }
}

// ============================================================================
// Kernel 2: f1[h,n] = Wh[h,n,:]·a1[h],  f2[h,n] = Wh[h,n,:]·a2[h]
// ============================================================================
__global__ void __launch_bounds__(256)
gat_f12(const float* __restrict__ Wh,   // (H*NN, 64)
        const float* __restrict__ a,    // (H, 128): [0:64]=a1, [64:128]=a2
        float* __restrict__ f1, float* __restrict__ f2, int total)
{
  const int idx = blockIdx.x * 256 + threadIdx.x;
  if (idx >= total) return;
  const int h = idx >> 12;                 // NN == 4096
  const float* wr = Wh + (size_t)idx * 64;
  const float* a1 = a + (size_t)h * 128;
  const float* a2 = a1 + 64;
  float s1 = 0.f, s2 = 0.f;
  #pragma unroll 8
  for (int o = 0; o < 64; ++o) { const float v = wr[o]; s1 += v * a1[o]; s2 += v * a2[o]; }
  f1[idx] = s1;
  f2[idx] = s2;
}

// ============================================================================
// Per-m-tile flash-attention step (shared by both attn kernels).
// Element order j=0..15 corresponds to the f16 16x32 A-operand layout for
// lane L (M=L&15, hi=L>>4):  col(j<8) = mt+8*hi+j ; col(j>=8) = mt+16+8*hi+(j-8)
// ============================================================================
__device__ __forceinline__ void
attn_tile_step(const int mk[16], const float f2v[16], float f1r,
               int M, int hi, int mt,
               float& mrun, float& lrun, v8f acc[4],
               const _Float16* __restrict__ Wt /* (64, NN) head base */)
{
  float e[16];
  float tmax = NEGBIG;
  #pragma unroll
  for (int j = 0; j < 16; ++j) {
    float g = f1r + f2v[j];
    g = g > 0.f ? g : 0.2f * g;                       // leaky relu
    const float ej = mk[j] > 0 ? g : NEGBIG;          // adjacency mask
    e[j] = ej;
    tmax = fmaxf(tmax, ej);
  }
  tmax = fmaxf(tmax, __shfl_xor(tmax, 16, 32));       // join the 2 row halves
  const float mnew  = fmaxf(mrun, tmax);
  const float scale = __expf(mrun - mnew);

  v16h P;
  float rs = 0.f;
  #pragma unroll
  for (int j = 0; j < 16; ++j) {
    const float pv = mk[j] > 0 ? __expf(e[j] - mnew) : 0.f;
    rs += pv;
    P[j] = (_Float16)pv;
  }
  rs += __shfl_xor(rs, 16, 32);
  lrun = lrun * scale + rs;
  mrun = mnew;

  #pragma unroll
  for (int c = 0; c < 8; ++c) {                       // C-layout row = c + 8*hi
    const float sc = __shfl(scale, c + 8 * hi, 32);
    acc[0][c] *= sc; acc[1][c] *= sc; acc[2][c] *= sc; acc[3][c] *= sc;
  }

  const int coff = mt + 8 * hi;
  #pragma unroll
  for (int ch = 0; ch < 4; ++ch) {
    const _Float16* Wtc = Wt + (size_t)(ch * 16 + M) * NN + coff;
    const v8h lo = *(const v8h*)(Wtc);                // halves K=8hi..8hi+7
    const v8h hh = *(const v8h*)(Wtc + 16);           // halves K=16+8hi..+7
    const v16h B = __builtin_shufflevector(lo, hh, 0, 1, 2, 3, 4, 5, 6, 7,
                                           8, 9, 10, 11, 12, 13, 14, 15);
    acc[ch] = __builtin_amdgcn_wmma_f32_16x16x32_f16(false, P, false, B,
                                                     (short)0, acc[ch], false, false);
  }
}

// ============================================================================
// Kernel 3a (layer 1): 8 waves per block = 8 heads over the SAME 16-row tile.
// The 16x32 adj tile is staged once per block into LDS with CDNA5 async
// loads (global_load_async_to_lds_b64, ASYNCcnt), double-buffered.
// ============================================================================
__global__ void __launch_bounds__(256)
gat_attn_l1(const float* __restrict__ f1,     // (8, NN)
            const float* __restrict__ f2,     // (8, NN)
            const _Float16* __restrict__ WhT, // (8, 64, NN)
            const int* __restrict__ adj,      // (NN, NN)
            float* __restrict__ out)          // (NN, 512), ELU applied
{
  __shared__ __align__(16) int tile[2][16 * 32];     // 2 x 2KB double buffer

  const int nt   = blockIdx.x;
  const int h    = threadIdx.x >> 5;                 // wave = head
  const int lane = threadIdx.x & 31;
  const int M    = lane & 15;
  const int hi   = lane >> 4;
  const int row  = nt * 16 + M;

  const float     f1r = f1[(size_t)h * NN + row];
  const float*    f2h = f2 + (size_t)h * NN;
  const _Float16* Wt  = WhT + (size_t)h * 64 * NN;

  // Per-lane async-copy slot: 256 lanes x 8B = one 16x32 int tile.
  const int rr = threadIdx.x >> 4;                   // 0..15 (tile row)
  const int cc = (threadIdx.x & 15) << 1;            // 0..30 (tile col pair)
  const int* gsrc = adj + (size_t)(nt * 16 + rr) * NN + cc;
  const unsigned ldsbase =
      (unsigned)(unsigned long long)(void*)&tile[0][0] + (unsigned)((rr * 32 + cc) * 4);

  #define ISSUE_TILE(buf, mt_next)                                              \
    do {                                                                        \
      const int* _s = gsrc + (mt_next);                                         \
      unsigned _d = ldsbase + (unsigned)((buf) * 2048);                         \
      asm volatile("global_load_async_to_lds_b64 %0, %1, off"                   \
                   :: "v"(_d), "v"(_s) : "memory");                             \
    } while (0)

  float mrun = NEGBIG, lrun = 0.f;
  v8f acc[4] = {v8f{}, v8f{}, v8f{}, v8f{}};

  ISSUE_TILE(0, 0);
  asm volatile("s_wait_asynccnt 0x0" ::: "memory");
  __syncthreads();

  for (int mt = 0; mt < NN; mt += 32) {
    const int cur = (mt >> 5) & 1;
    if (mt + 32 < NN) ISSUE_TILE(cur ^ 1, mt + 32);

    // adj from LDS (two 16B + two 16B per lane), f2 from global
    const int* tb = &tile[cur][M * 32 + 8 * hi];
    const int4 A0 = *(const int4*)(tb);
    const int4 A1 = *(const int4*)(tb + 4);
    const int4 A2 = *(const int4*)(tb + 16);
    const int4 A3 = *(const int4*)(tb + 20);
    const int mk[16] = {A0.x, A0.y, A0.z, A0.w, A1.x, A1.y, A1.z, A1.w,
                        A2.x, A2.y, A2.z, A2.w, A3.x, A3.y, A3.z, A3.w};
    const float* fb = f2h + mt + 8 * hi;
    const float4 F0 = *(const float4*)(fb);
    const float4 F1 = *(const float4*)(fb + 4);
    const float4 F2 = *(const float4*)(fb + 16);
    const float4 F3 = *(const float4*)(fb + 20);
    const float f2v[16] = {F0.x, F0.y, F0.z, F0.w, F1.x, F1.y, F1.z, F1.w,
                           F2.x, F2.y, F2.z, F2.w, F3.x, F3.y, F3.z, F3.w};

    attn_tile_step(mk, f2v, f1r, M, hi, mt, mrun, lrun, acc, Wt);

    asm volatile("s_wait_asynccnt 0x0" ::: "memory");  // next tile landed
    __syncthreads();                                   // everyone done with cur
  }
  #undef ISSUE_TILE

  #pragma unroll
  for (int c = 0; c < 8; ++c) {
    const float l   = __shfl(lrun, c + 8 * hi, 32);
    const float inv = 1.0f / l;
    const int orow  = nt * 16 + c + 8 * hi;
    float* orw = out + (size_t)orow * 512 + h * 64;
    #pragma unroll
    for (int ch = 0; ch < 4; ++ch) {
      float vv = acc[ch][c] * inv;
      vv = vv > 0.f ? vv : (__expf(vv) - 1.f);         // ELU
      orw[ch * 16 + M] = vv;
    }
  }
}

// ============================================================================
// Kernel 3b (layer 2): split-m flash attention.  8 waves/block on distinct
// n-tiles; grid.y = m-segment.  Emits partial (acc, m, l) per segment.
// ============================================================================
__global__ void __launch_bounds__(256)
gat_attn_split(const float* __restrict__ f1,     // (NN)
               const float* __restrict__ f2,     // (NN)
               const _Float16* __restrict__ WhT, // (64, NN)
               const int* __restrict__ adj,      // (NN, NN)
               float* __restrict__ Pacc,         // (S, NN, 64)
               float* __restrict__ Pm,           // (S, NN)
               float* __restrict__ Pl,           // (S, NN)
               int seglen)
{
  const int wave = threadIdx.x >> 5;
  const int lane = threadIdx.x & 31;
  const int ntile = blockIdx.x * 8 + wave;
  const int seg   = blockIdx.y;
  const int m0    = seg * seglen;
  const int M  = lane & 15;
  const int hi = lane >> 4;
  const int row = ntile * 16 + M;

  const float f1r = f1[row];
  const int* adjrow = adj + (size_t)row * NN;

  float mrun = NEGBIG, lrun = 0.f;
  v8f acc[4] = {v8f{}, v8f{}, v8f{}, v8f{}};

  for (int mt = m0; mt < m0 + seglen; mt += 32) {
    __builtin_prefetch(adjrow + mt + 256, 0, 0);       // global_prefetch_b8

    const int* ab = adjrow + mt + 8 * hi;
    const int4 A0 = *(const int4*)(ab);
    const int4 A1 = *(const int4*)(ab + 4);
    const int4 A2 = *(const int4*)(ab + 16);
    const int4 A3 = *(const int4*)(ab + 20);
    const int mk[16] = {A0.x, A0.y, A0.z, A0.w, A1.x, A1.y, A1.z, A1.w,
                        A2.x, A2.y, A2.z, A2.w, A3.x, A3.y, A3.z, A3.w};
    const float* fb = f2 + mt + 8 * hi;
    const float4 F0 = *(const float4*)(fb);
    const float4 F1 = *(const float4*)(fb + 4);
    const float4 F2 = *(const float4*)(fb + 16);
    const float4 F3 = *(const float4*)(fb + 20);
    const float f2v[16] = {F0.x, F0.y, F0.z, F0.w, F1.x, F1.y, F1.z, F1.w,
                           F2.x, F2.y, F2.z, F2.w, F3.x, F3.y, F3.z, F3.w};

    attn_tile_step(mk, f2v, f1r, M, hi, mt, mrun, lrun, acc, WhT);
  }

  float* pa = Pacc + (size_t)seg * NN * 64;
  #pragma unroll
  for (int c = 0; c < 8; ++c) {
    const int orow = ntile * 16 + c + 8 * hi;
    #pragma unroll
    for (int ch = 0; ch < 4; ++ch)
      pa[(size_t)orow * 64 + ch * 16 + M] = acc[ch][c];
  }
  if (hi == 0) {                       // lanes 0..15 hold row stats
    Pm[(size_t)seg * NN + row] = mrun;
    Pl[(size_t)seg * NN + row] = lrun;
  }
}

// ============================================================================
// Kernel 4: merge m-segments with log-sum-exp weights, normalize, ELU.
// ============================================================================
__global__ void __launch_bounds__(256)
gat_combine(const float* __restrict__ Pacc, const float* __restrict__ Pm,
            const float* __restrict__ Pl, float* __restrict__ out, int S)
{
  const int idx = blockIdx.x * 256 + threadIdx.x;     // n*64 + o
  const int n = idx >> 6;
  float mmax = NEGBIG;
  for (int s = 0; s < S; ++s) mmax = fmaxf(mmax, Pm[(size_t)s * NN + n]);
  float num = 0.f, den = 0.f;
  for (int s = 0; s < S; ++s) {
    const float w = __expf(Pm[(size_t)s * NN + n] - mmax);
    num += w * Pacc[(size_t)s * NN * 64 + idx];
    den += w * Pl[(size_t)s * NN + n];
  }
  float vv = num / den;
  vv = vv > 0.f ? vv : (__expf(vv) - 1.f);            // ELU
  out[idx] = vv;
}

// ============================================================================
extern "C" void kernel_launch(void* const* d_in, const int* in_sizes, int n_in,
                              void* d_out, int out_size, void* d_ws, size_t ws_size,
                              hipStream_t stream)
{
  const float* x   = (const float*)d_in[0];   // (4096, 128)
  const int*   adj = (const int*)d_in[1];     // (4096, 4096)
  // d_in[2] = observation (unused by the reference computation)
  const float* Whd = (const float*)d_in[3];   // (8, 128, 64)
  const float* ah  = (const float*)d_in[4];   // (8, 128, 1)
  const float* Wo  = (const float*)d_in[5];   // (512, 64)
  const float* ao  = (const float*)d_in[6];   // (128, 1)

  char* ws = (char*)d_ws;
  size_t off = 0;
  auto alloc = [&](size_t bytes) -> void* {
    void* p = ws + off;
    off = (off + bytes + 255) & ~(size_t)255;
    return p;
  };

  const int S = 8;                              // layer-2 m-segments
  float*    Wh1  = (float*)   alloc((size_t)8 * NN * 64 * sizeof(float));
  _Float16* WhT1 = (_Float16*)alloc((size_t)8 * 64 * NN * sizeof(_Float16));
  float*    f1a  = (float*)   alloc((size_t)8 * NN * sizeof(float));
  float*    f2a  = (float*)   alloc((size_t)8 * NN * sizeof(float));
  float*    hbuf = (float*)   alloc((size_t)NN * 512 * sizeof(float));
  float*    Wh2  = (float*)   alloc((size_t)NN * 64 * sizeof(float));
  _Float16* WhT2 = (_Float16*)alloc((size_t)64 * NN * sizeof(_Float16));
  float*    f1b  = (float*)   alloc((size_t)NN * sizeof(float));
  float*    f2b  = (float*)   alloc((size_t)NN * sizeof(float));
  float*    Pacc = (float*)   alloc((size_t)S * NN * 64 * sizeof(float));
  float*    Pm   = (float*)   alloc((size_t)S * NN * sizeof(float));
  float*    Pl   = (float*)   alloc((size_t)S * NN * sizeof(float));

  // ---- Layer 1 (8 heads, F=128) ----
  gat_gemm_wh<<<dim3(NN / 16, 8), 32, 0, stream>>>(x, Whd, Wh1, WhT1, 128);
  gat_f12    <<<dim3(8 * NN / 256), 256, 0, stream>>>(Wh1, ah, f1a, f2a, 8 * NN);
  gat_attn_l1<<<dim3(NN / 16), 256, 0, stream>>>(f1a, f2a, WhT1, adj, hbuf);

  // ---- Layer 2 (1 head, F=512), split-m for occupancy ----
  gat_gemm_wh<<<dim3(NN / 16, 1), 32, 0, stream>>>(hbuf, Wo, Wh2, WhT2, 512);
  gat_f12    <<<dim3(NN / 256), 256, 0, stream>>>(Wh2, ao, f1b, f2b, NN);
  gat_attn_split<<<dim3(NN / 16 / 8, S), 256, 0, stream>>>(f1b, f2b, WhT2, adj,
                                                           Pacc, Pm, Pl, NN / S);
  gat_combine<<<dim3(NN * 64 / 256), 256, 0, stream>>>(Pacc, Pm, Pl, (float*)d_out, S);
}
